// GNN_62440234549283
// MI455X (gfx1250) — matrix operand order repaired and checked by reading, not compiled, not run on hardware
//
#include <hip/hip_runtime.h>
#include <hip/hip_bf16.h>
#include <math.h>

// ---------------------------------------------------------------------------
// MI455X (gfx1250) hetero-GATv2 forward.
//  * Dense projections: bf16 WMMA (v_wmma_f32_16x16x32_bf16), f32 accumulate.
//    K-step 64 -> two WMMAs per barrier pair; all M are multiples of 16 so
//    the hot loop has no guards at all; K remainder split out.
//  * Edge softmax: 3-pass float-atomic pipeline (L2-resident accumulators).
//  * Relation attention: scalar kernels (tiny FLOP count).
// ---------------------------------------------------------------------------

typedef __attribute__((ext_vector_type(16))) __bf16 v16bf;
typedef __attribute__((ext_vector_type(8)))  float  v8f;

__device__ __forceinline__ unsigned bfbits(float f) {
  union { float f; unsigned u; } c; c.f = f;
  unsigned u = c.u;
  u += 0x7FFFu + ((u >> 16) & 1u);            // round to nearest even
  return u >> 16;
}
__device__ __forceinline__ unsigned pack2(float a, float b) {
  return bfbits(a) | (bfbits(b) << 16);
}
__device__ __forceinline__ __bf16 bfval(float f) {
  unsigned short s = (unsigned short)bfbits(f);
  return *(__bf16*)&s;
}

union FragBF { v16bf v; unsigned u[8]; };

// ---------------------------------------------------------------------------
// C[M,128] = A[M,K] @ W[K,128] + bias.  One block = 16 rows x 128 cols,
// 8 waves, each wave owns one 16x16 output tile; K-step 64 = 2 WMMAs/iter.
// LDS tiles are PAIR-PACKED: one u32 = {bf16 k_even, bf16 k_odd}, exactly the
// 16-bit WMMA VGPR packing (ISA 7.12.2) -> every fragment reg is one
// ds_load_b32 (fused by the compiler into ds_load_2addr_b32 pairs).
// REQUIRES: M % 16 == 0 (true for every call in this pipeline).
// ---------------------------------------------------------------------------
__global__ __launch_bounds__(256)
void gemm_bias_wmma(const float* __restrict__ A, int M, int K,
                    const float* __restrict__ W,      // [K,128] row major
                    const float* __restrict__ bias,   // [128]
                    float* __restrict__ out, int ostride)
{
  __shared__ unsigned Apk[16][33];    // [m][k/2] for 64 k, +1 u32 pad
  __shared__ unsigned Bpk[32][129];   // [k/2][n] for 64 k, +1 u32 pad
  const int tid  = threadIdx.x;
  const int wave = tid >> 5;
  const int lane = tid & 31;
  const int m0   = blockIdx.x * 16;
  const int n0   = wave * 16;
  const int mrow = lane & 15;
  const int hi   = lane >> 4;
  // A cooperative-load coords: one packed u32 per thread per 32-k sub-step
  const int am = tid >> 4, akp = tid & 15;
  const float* Arow = A + (long long)(m0 + am) * K + 2 * akp;

  v8f acc = {};
  const int kfull = K & ~63;
  int k0 = 0;
  for (; k0 < kfull; k0 += 64) {
    // ---- A tile (16x64): unguarded, 2 packed u32 per thread
    #pragma unroll
    for (int sub = 0; sub < 2; ++sub) {
      int ks = k0 + 32 * sub;
      Apk[am][akp + 16 * sub] = pack2(Arow[ks], Arow[ks + 1]);
    }
    // ---- B tile (64x128): 8 x global_load_b128 per thread, coalesced
    #pragma unroll
    for (int i = 0; i < 8; ++i) {
      int idx = tid + i * 256;                    // 0..2047
      int r = idx >> 5;                           // k row 0..63
      int n = (idx & 31) * 4;                     // col 0..124
      const float4 w4 = *(const float4*)(W + (long long)(k0 + r) * 128 + n);
      __bf16* p = (__bf16*)(&Bpk[r >> 1][n]) + (r & 1);
      p[0] = bfval(w4.x); p[2] = bfval(w4.y); p[4] = bfval(w4.z); p[6] = bfval(w4.w);
    }
    __syncthreads();
    #pragma unroll
    for (int sub = 0; sub < 2; ++sub) {
      FragBF af, bf;
      #pragma unroll
      for (int v = 0; v < 8; ++v) {
        int kpA = (v < 4 ? v : v + 4) + hi * 4;   // A 16x32 layout, pair index
        af.u[v] = Apk[mrow][kpA + 16 * sub];
        bf.u[v] = Bpk[16 * sub + kpA][n0 + mrow];   // B 32x16 layout
      }
      acc = __builtin_amdgcn_wmma_f32_16x16x32_bf16(
          false, af.v, false, bf.v, (short)0, acc, false, false);
    }
    __syncthreads();
  }
  for (; k0 < K; k0 += 32) {                      // K remainder (e.g. 148)
    int gk = k0 + 2 * akp;
    float f0 = (gk     < K) ? Arow[k0]     : 0.f;
    float f1 = (gk + 1 < K) ? Arow[k0 + 1] : 0.f;
    Apk[am][akp] = pack2(f0, f1);
    #pragma unroll
    for (int i = 0; i < 4; ++i) {
      int idx = tid + i * 256;
      int r = idx >> 5;
      int n = (idx & 31) * 4;
      float4 w4 = make_float4(0.f, 0.f, 0.f, 0.f);
      if (k0 + r < K) w4 = *(const float4*)(W + (long long)(k0 + r) * 128 + n);
      __bf16* p = (__bf16*)(&Bpk[r >> 1][n]) + (r & 1);
      p[0] = bfval(w4.x); p[2] = bfval(w4.y); p[4] = bfval(w4.z); p[6] = bfval(w4.w);
    }
    __syncthreads();
    FragBF af, bf;
    #pragma unroll
    for (int v = 0; v < 8; ++v) {
      int kpA = (v < 4 ? v : v + 4) + hi * 4;
      af.u[v] = Apk[mrow][kpA];
      bf.u[v] = Bpk[kpA][n0 + mrow];
    }
    acc = __builtin_amdgcn_wmma_f32_16x16x32_bf16(
        false, af.v, false, bf.v, (short)0, acc, false, false);
    __syncthreads();
  }
  // C/D layout: VGPR r -> M = r + hi*8, N = lane&15; unguarded (M%16==0)
  #pragma unroll
  for (int r = 0; r < 8; ++r)
    out[(long long)(m0 + r + hi * 8) * ostride + n0 + mrow] = acc[r] + bias[n0 + mrow];
}

// ---------------------------------------------------------------------------
__global__ void fill_kernel(float* __restrict__ p, float v, long long n) {
  long long i = (long long)blockIdx.x * blockDim.x + threadIdx.x;
  if (i < n) p[i] = v;
}

__device__ __forceinline__ void atomicMaxFloat(float* addr, float v) {
  if (v >= 0.f) atomicMax((int*)addr, __float_as_int(v));
  else          atomicMin((unsigned int*)addr, __float_as_uint(v));
}

// pass 1: e[e,h] = attn_h . leaky_relu(hs[src]+hd[dst]); atomic max per (dst,h)
__global__ void edge_logits_max(const int* __restrict__ src, const int* __restrict__ dst,
                                const float* __restrict__ hs, const float* __restrict__ hd,
                                const float* __restrict__ attn,   // [8*16]
                                float* __restrict__ ebuf, float* __restrict__ maxb, int E)
{
  int t = blockIdx.x * blockDim.x + threadIdx.x;
  if (t >= E * 8) return;
  int e = t >> 3, h = t & 7;
  int s = src[e], d = dst[e];
  const float4* ps = (const float4*)(hs + (long long)s * 128 + h * 16);
  const float4* pd = (const float4*)(hd + (long long)d * 128 + h * 16);
  const float4* pa = (const float4*)(attn + h * 16);
  float acc = 0.f;
  #pragma unroll
  for (int j = 0; j < 4; ++j) {
    float4 a = ps[j], b = pd[j], w = pa[j];
    float m;
    m = a.x + b.x; m = (m > 0.f) ? m : 0.2f * m; acc += m * w.x;
    m = a.y + b.y; m = (m > 0.f) ? m : 0.2f * m; acc += m * w.y;
    m = a.z + b.z; m = (m > 0.f) ? m : 0.2f * m; acc += m * w.z;
    m = a.w + b.w; m = (m > 0.f) ? m : 0.2f * m; acc += m * w.w;
  }
  ebuf[t] = acc;
  atomicMaxFloat(&maxb[d * 8 + h], acc);
}

// pass 2: a = exp(e - max); atomic sum per (dst,h)
__global__ void edge_exp_sum(const int* __restrict__ dst,
                             float* __restrict__ ebuf,
                             const float* __restrict__ maxb,
                             float* __restrict__ sumb, int E)
{
  int t = blockIdx.x * blockDim.x + threadIdx.x;
  if (t >= E * 8) return;
  int e = t >> 3, h = t & 7;
  int d = dst[e];
  float a = expf(ebuf[t] - maxb[d * 8 + h]);
  ebuf[t] = a;
  atomicAdd(&sumb[d * 8 + h], a);
}

// pass 3: out[dst,:] += (a/sum)*ew * hs[src,:]; 4 columns per thread
__global__ void edge_scatter(const int* __restrict__ src, const int* __restrict__ dst,
                             const float* __restrict__ ebuf, const float* __restrict__ sumb,
                             const float* __restrict__ ew, const float* __restrict__ hs,
                             float* __restrict__ outp, int ostride, int E)
{
  long long t = (long long)blockIdx.x * blockDim.x + threadIdx.x;
  if (t >= (long long)E * 32) return;
  int e = (int)(t >> 5), q = (int)(t & 31);
  int dcol = q * 4, h = q >> 2;
  int s = src[e], d = dst[e];
  float coef = ebuf[e * 8 + h] / sumb[d * 8 + h] * ew[e];
  float4 v = *(const float4*)(hs + (long long)s * 128 + dcol);
  float* o = outp + (long long)d * ostride + dcol;
  atomicAdd(o + 0, coef * v.x);
  atomicAdd(o + 1, coef * v.y);
  atomicAdd(o + 2, coef * v.z);
  atomicAdd(o + 3, coef * v.w);
}

// ---------------------------------------------------------------------------
// relation attention: w[n,r] = tanh(z[n,r]@P1 + b1)@P2 ; rsums[r] += w
__global__ void relatt_score(const float* z0, int s0, const float* z1, int s1,
                             const float* z2, int s2, int R, int N, int do_relu,
                             const float* __restrict__ P1,   // [128,16]
                             const float* __restrict__ b1,   // [16]
                             const float* __restrict__ P2,   // [16]
                             float* __restrict__ rsums)
{
  int t = blockIdx.x * blockDim.x + threadIdx.x;
  if (t >= N * R) return;
  int r = t / N, n = t - r * N;
  const float* z  = (r == 0) ? z0 : (r == 1) ? z1 : z2;
  int st          = (r == 0) ? s0 : (r == 1) ? s1 : s2;
  const float* row = z + (long long)n * st;
  float tj[16];
  #pragma unroll
  for (int j = 0; j < 16; ++j) tj[j] = b1[j];
  for (int d = 0; d < 128; ++d) {
    float v = row[d];
    if (do_relu) v = v > 0.f ? v : 0.f;
    #pragma unroll
    for (int j = 0; j < 16; ++j) tj[j] += v * P1[d * 16 + j];
  }
  float w = 0.f;
  #pragma unroll
  for (int j = 0; j < 16; ++j) w += tanhf(tj[j]) * P2[j];
  atomicAdd(&rsums[r], w);
}

__global__ void beta_kernel(const float* __restrict__ rsums, float* __restrict__ beta,
                            int R, float invN)
{
  if (threadIdx.x == 0 && blockIdx.x == 0) {
    float m = -3.0e38f;
    for (int r = 0; r < R; ++r) { float v = rsums[r] * invN; if (v > m) m = v; }
    float e[8], s = 0.f;
    for (int r = 0; r < R; ++r) { e[r] = expf(rsums[r] * invN - m); s += e[r]; }
    for (int r = 0; r < R; ++r) beta[r] = e[r] / s;
  }
}

__global__ void relatt_combine(const float* z0, int s0, const float* z1, int s1,
                               const float* z2, int s2, int R, int N, int do_relu,
                               const float* __restrict__ beta,
                               float* __restrict__ out, int ostride)
{
  long long t = (long long)blockIdx.x * blockDim.x + threadIdx.x;
  if (t >= (long long)N * 128) return;
  int n = (int)(t >> 7), dcol = (int)(t & 127);
  float acc = 0.f;
  for (int r = 0; r < R; ++r) {
    const float* z = (r == 0) ? z0 : (r == 1) ? z1 : z2;
    int st         = (r == 0) ? s0 : (r == 1) ? s1 : s2;
    float v = z[(long long)n * st + dcol];
    if (do_relu) v = v > 0.f ? v : 0.f;
    acc += beta[r] * v;
  }
  out[(long long)n * ostride + dcol] = acc;
}

// CAT[n,0:128] = last[:,1,:] (strided in d_out), CAT[n,128:256] = total_ingre_emb
__global__ void concat_kernel(const float* __restrict__ last, const float* __restrict__ tie,
                              float* __restrict__ cat, int N)
{
  long long t = (long long)blockIdx.x * blockDim.x + threadIdx.x;
  if (t >= (long long)N * 256) return;
  int n = (int)(t >> 8), dcol = (int)(t & 255);
  cat[t] = (dcol < 128) ? last[(long long)n * 256 + 128 + dcol]
                        : tie[(long long)n * 128 + (dcol - 128)];
}

// ---------------------------------------------------------------------------
extern "C" void kernel_launch(void* const* d_in, const int* in_sizes, int n_in,
                              void* d_out, int out_size, void* d_ws, size_t ws_size,
                              hipStream_t stream)
{
  (void)in_sizes; (void)n_in; (void)out_size; (void)ws_size;
  const int NC = 10000, NI = 20000, NR = 20000, E = 400000;
  const float NEG_INF = -3.0e38f;

  // ---- input indices (setup_inputs insertion order; params = jax tree
  //      leaves, dict keys sorted at each level) ----
  enum {
    I_FC = 0, I_FI, I_FR, I_TIE, I_PIC,
    I_S_CI = 5,  I_D_CI,  I_W_CI,
    I_S_IC = 8,  I_D_IC,  I_W_IC,     // 'ic' unused by the forward pass
    I_S_IR = 11, I_D_IR,  I_W_IR,
    I_S_RI = 14, I_D_RI,  I_W_RI,
    I_S_RR = 17, I_D_RR,  I_W_RR,
    I_S_II = 20, I_D_II,  I_W_II,
    I_S2_IR = 23, I_D2_IR, I_W2_IR,
    I_S2_RR = 26, I_D2_RR, I_W2_RR,
    P_COMB_W = 29, P_COMB_B,
    P_EMB_W = 31,  P_EMB_B,
    P_G1_CI = 33,                     // each gat block: Wd, Ws, attn, bd, bs
    P_G1_IC = 38, P_G1_II = 43, P_G1_IR = 48, P_G1_RI = 53, P_G1_RR = 58,
    P_G2_IR = 63, P_G2_RR = 68,
    P_RA_P1 = 73, P_RA_P2 = 74, P_RA_B1 = 75
  };
  auto F = [&](int i) { return (const float*)d_in[i]; };
  auto I = [&](int i) { return (const int*)d_in[i]; };

  // ---- workspace layout (floats) ----
  float* ws   = (float*)d_ws;
  float* HS   = ws;                          //  20000*128
  float* HD   = ws + 2560000;                //  20000*128
  float* EB   = ws + 5120000;                //  400000*8
  float* MAXB = ws + 8320000;                //  20000*8
  float* SUMB = ws + 8480000;                //  20000*8
  float* AGG[5];
  for (int i = 0; i < 5; ++i) AGG[i] = ws + 8640000 + (long long)i * 2560000;
  float* H_R    = ws + 21440000;             //  20000*128
  float* TEMP   = ws + 24000000;             //  20000*128
  float* CAT    = ws + 26560000;             //  20000*256
  float* RECPRE = ws + 31680000;             //  20000*128
  float* RSUM   = ws + 34240000;             //  8
  float* BETA   = RSUM + 8;                  //  4

  float* OUT  = (float*)d_out;
  float* REC  = OUT;                         // [NR,128]
  float* HIo  = OUT + 2560000;               // [NI,128]  (returned h_i)
  float* LAST = OUT + 5120000;               // [NR,2,128] interleaved

  auto fill = [&](float* p, float v, long long n) {
    fill_kernel<<<dim3((unsigned)((n + 255) / 256)), dim3(256), 0, stream>>>(p, v, n);
  };
  auto gemm = [&](const float* A, int M, int K, const float* W, const float* b,
                  float* out, int ostride) {
    gemm_bias_wmma<<<dim3(M / 16), dim3(256), 0, stream>>>(A, M, K, W, b, out, ostride);
  };
  // one GATv2 relation: project, edge softmax, scatter into outp (pre-zeroed)
  auto run_gat = [&](const float* xs, int Ns, const float* xd, int Nd, int K,
                     const int* src, const int* dst, const float* ew,
                     const float* Wd, const float* Ws, const float* attn,
                     const float* bd, const float* bs,
                     float* outp, int ostride) {
    gemm(xs, Ns, K, Ws, bs, HS, 128);
    gemm(xd, Nd, K, Wd, bd, HD, 128);
    fill(MAXB, NEG_INF, (long long)Nd * 8);
    fill(SUMB, 0.f,     (long long)Nd * 8);
    int g8 = (E * 8 + 255) / 256;
    edge_logits_max<<<dim3(g8), dim3(256), 0, stream>>>(src, dst, HS, HD, attn, EB, MAXB, E);
    edge_exp_sum  <<<dim3(g8), dim3(256), 0, stream>>>(dst, EB, MAXB, SUMB, E);
    long long n32 = (long long)E * 32;
    edge_scatter<<<dim3((unsigned)((n32 + 255) / 256)), dim3(256), 0, stream>>>(
        src, dst, EB, SUMB, ew, HS, outp, ostride, E);
  };
  auto gp = [&](int base, int off) { return F(base + off); };  // 0:Wd 1:Ws 2:attn 3:bd 4:bs

  // ================= layer 1 (relations feeding ingredient / recipe) =======
  struct L1 { const float* xs; int Ns; const float* xd; int Nd;
              int e; int p; int agg; };
  L1 rels[5] = {
    { F(I_FC), NC, F(I_FI), NI, I_S_CI, P_G1_CI, 0 },   // ci -> i
    { F(I_FI), NI, F(I_FI), NI, I_S_II, P_G1_II, 1 },   // ii -> i
    { F(I_FR), NR, F(I_FI), NI, I_S_RI, P_G1_RI, 2 },   // ri -> i
    { F(I_FI), NI, F(I_FR), NR, I_S_IR, P_G1_IR, 3 },   // ir -> r
    { F(I_FR), NR, F(I_FR), NR, I_S_RR, P_G1_RR, 4 },   // rr -> r
  };
  for (int k = 0; k < 5; ++k) {
    L1& r = rels[k];
    fill(AGG[r.agg], 0.f, (long long)r.Nd * 128);
    run_gat(r.xs, r.Ns, r.xd, r.Nd, 148,
            I(r.e), I(r.e + 1), F(r.e + 2),
            gp(r.p, 0), gp(r.p, 1), gp(r.p, 2), gp(r.p, 3), gp(r.p, 4),
            AGG[r.agg], 128);
  }

  const float* P1 = F(P_RA_P1); const float* P2 = F(P_RA_P2); const float* B1 = F(P_RA_B1);

  // h_i = rel_att(relu(stack(ci,ii,ri)))  -> d_out h_i region
  fill(RSUM, 0.f, 8);
  relatt_score<<<dim3((3 * NI + 255) / 256), dim3(256), 0, stream>>>(
      AGG[0], 128, AGG[1], 128, AGG[2], 128, 3, NI, 1, P1, B1, P2, RSUM);
  beta_kernel<<<dim3(1), dim3(32), 0, stream>>>(RSUM, BETA, 3, 1.0f / NI);
  relatt_combine<<<dim3((unsigned)(((long long)NI * 128 + 255) / 256)), dim3(256), 0, stream>>>(
      AGG[0], 128, AGG[1], 128, AGG[2], 128, 3, NI, 1, BETA, HIo, 128);

  // h_r = rel_att(relu(stack(ir,rr)))
  fill(RSUM, 0.f, 8);
  relatt_score<<<dim3((2 * NR + 255) / 256), dim3(256), 0, stream>>>(
      AGG[3], 128, AGG[4], 128, AGG[4], 128, 2, NR, 1, P1, B1, P2, RSUM);
  beta_kernel<<<dim3(1), dim3(32), 0, stream>>>(RSUM, BETA, 2, 1.0f / NR);
  relatt_combine<<<dim3((unsigned)(((long long)NR * 128 + 255) / 256)), dim3(256), 0, stream>>>(
      AGG[3], 128, AGG[4], 128, AGG[4], 128, 2, NR, 1, BETA, H_R, 128);

  // ================= layer 2: ir2 -> last[:,0,:], rr2 -> last[:,1,:] =======
  fill(LAST, 0.f, (long long)NR * 256);
  run_gat(HIo, NI, H_R, NR, 128,
          I(I_S2_IR), I(I_D2_IR), F(I_W2_IR),
          gp(P_G2_IR, 0), gp(P_G2_IR, 1), gp(P_G2_IR, 2), gp(P_G2_IR, 3), gp(P_G2_IR, 4),
          LAST, 256);
  run_gat(H_R, NR, H_R, NR, 128,
          I(I_S2_RR), I(I_D2_RR), F(I_W2_RR),
          gp(P_G2_RR, 0), gp(P_G2_RR, 1), gp(P_G2_RR, 2), gp(P_G2_RR, 3), gp(P_G2_RR, 4),
          LAST + 128, 256);

  // temp = [last[:,1,:] | total_ingre_emb] @ comb.W + comb.b
  concat_kernel<<<dim3((unsigned)(((long long)NR * 256 + 255) / 256)), dim3(256), 0, stream>>>(
      LAST, F(I_TIE), CAT, NR);
  gemm(CAT, NR, 256, F(P_COMB_W), F(P_COMB_B), TEMP, 128);

  // rec = rel_att(stack(last0, temp, PIC)) @ emb.W + emb.b   (no relu here)
  fill(RSUM, 0.f, 8);
  relatt_score<<<dim3((3 * NR + 255) / 256), dim3(256), 0, stream>>>(
      LAST, 256, TEMP, 128, F(I_PIC), 128, 3, NR, 0, P1, B1, P2, RSUM);
  beta_kernel<<<dim3(1), dim3(32), 0, stream>>>(RSUM, BETA, 3, 1.0f / NR);
  relatt_combine<<<dim3((unsigned)(((long long)NR * 128 + 255) / 256)), dim3(256), 0, stream>>>(
      LAST, 256, TEMP, 128, F(I_PIC), 128, 3, NR, 0, BETA, RECPRE, 128);
  gemm(RECPRE, NR, 128, F(P_EMB_W), F(P_EMB_B), REC, 128);
}
